// EW_Conv_88880053223550
// MI455X (gfx1250) — matrix operand order repaired and checked by reading, not compiled, not run on hardware
//
#include <hip/hip_runtime.h>

#define N_NODES 50000
#define N_EDGES 625000
#define D 128

typedef __attribute__((ext_vector_type(16))) _Float16 v16h;
typedef __attribute__((ext_vector_type(8)))  float    v8f;

// ---------------- kernel 1: zero the workspace (agg [N,128] f32, deg [N] i32) ----
__global__ void zero_ws_kernel(float4* __restrict__ agg4, int* __restrict__ deg,
                               int n4, int n) {
    int i = blockIdx.x * blockDim.x + threadIdx.x;
    if (i < n4) {
        float4 z; z.x = 0.f; z.y = 0.f; z.z = 0.f; z.w = 0.f;
        agg4[i] = z;
    }
    if (i < n) deg[i] = 0;
}

// ---------------- kernel 2: edge gather + scatter-add --------------------------
// One wave32 per edge; lane l owns 4 consecutive features (float4).
__global__ __launch_bounds__(256)
void edge_kernel(const float* __restrict__ h, const float* __restrict__ d,
                 const int* __restrict__ src, const int* __restrict__ dst,
                 float* __restrict__ agg, int* __restrict__ deg) {
    int lane = threadIdx.x & 31;
    int e = blockIdx.x * 8 + (threadIdx.x >> 5);
    if (e >= N_EDGES) return;

    int s = src[e];
    int t = dst[e];
    float w = 1.0f - d[e];

    const float4 hs = *((const float4*)(h + (size_t)s * D) + lane);
    const float4 hd = *((const float4*)(h + (size_t)t * D) + lane);

    float4 m;
    m.x = fmaf(hs.x, w, hd.x);
    m.y = fmaf(hs.y, w, hd.y);
    m.z = fmaf(hs.z, w, hd.z);
    m.w = fmaf(hs.w, w, hd.w);

    float* ap = agg + (size_t)t * D + lane * 4;
    unsafeAtomicAdd(ap + 0, m.x);   // -> global_atomic_add_f32 (L2-resident)
    unsafeAtomicAdd(ap + 1, m.y);
    unsafeAtomicAdd(ap + 2, m.z);
    unsafeAtomicAdd(ap + 3, m.w);

    if (lane == 0) atomicAdd(deg + t, 1);
}

// ---------------- kernel 3: relu(h_new @ W^T + b) via split-precision WMMA -----
// Block = 256 threads = 8 waves. Block owns 16 rows (M tile); wave w owns
// output columns [16w, 16w+16). K = 128 done as 4 chunks of 32 with
// 3 WMMAs per chunk (hi*hi + hi*lo + lo*hi) for ~fp32 accuracy.
__global__ __launch_bounds__(256)
void gemm_kernel(const float* __restrict__ h, const float* __restrict__ agg,
                 const int* __restrict__ deg, const float* __restrict__ W,
                 const float* __restrict__ b, float* __restrict__ out) {
    __shared__ float As[16][132];   // 16 rows x 128 (+4 pad vs 64-bank LDS)

    const int mtile = blockIdx.x;
    const int t = threadIdx.x;

    // Stage h_new tile (deg>0 ? agg : h) into LDS: 16 rows x 128, 8 floats/thread
    {
        int r  = t >> 4;            // 0..15
        int c  = (t & 15) * 8;      // 0..120
        int rg = mtile * 16 + r;
        const float* srcp = (deg[rg] > 0) ? (agg + (size_t)rg * D)
                                          : (h   + (size_t)rg * D);
        float4 v0 = *(const float4*)(srcp + c);
        float4 v1 = *(const float4*)(srcp + c + 4);
        *(float4*)&As[r][c]     = v0;
        *(float4*)&As[r][c + 4] = v1;
    }
    __syncthreads();

    const int lane = t & 31;
    const int g    = lane >> 4;       // lane group (0/1)
    const int rl   = lane & 15;       // A row / B,C column within tile
    const int wid  = t >> 5;          // wave id = N tile
    const int ng   = wid * 16 + rl;   // global output column

    v8f acc = {};
    const float* wrow = W + (size_t)ng * D;   // B[k][n] = W[n][k]

#pragma unroll
    for (int kc = 0; kc < 4; ++kc) {
        // ---- A fragment: 16x32 f16; lane row=rl, halves 0..7 -> K=ka..ka+7,
        //      halves 8..15 -> K=ka+16..ka+23, ka = 32*kc + 8*g (ISA layout)
        const int ka = kc * 32 + 8 * g;
        float av[16];
#pragma unroll
        for (int i = 0; i < 8; ++i) av[i]     = As[rl][ka + i];
#pragma unroll
        for (int i = 0; i < 8; ++i) av[8 + i] = As[rl][ka + 16 + i];

        v16h ah, al;
#pragma unroll
        for (int i = 0; i < 16; ++i) {
            _Float16 hi = (_Float16)av[i];
            ah[i] = hi;
            al[i] = (_Float16)(av[i] - (float)hi);
        }

        // ---- B fragment: 32x16 f16; lane col=ng, half i -> K = 32*kc+8*g+i (i<8), 32*kc+8*g+16+(i-8) (i>=8)
        const float* wp = wrow + kc * 32 + 8 * g;
        v16h bh, bl;
#pragma unroll
        for (int i = 0; i < 16; ++i) {
            float bv = wp[i < 8 ? i : i + 8];
            _Float16 hi = (_Float16)bv;
            bh[i] = hi;
            bl[i] = (_Float16)(bv - (float)hi);
        }

        // split-precision accumulation: A*B ~= AhBh + AhBl + AlBh  (f32 acc)
        acc = __builtin_amdgcn_wmma_f32_16x16x32_f16(false, ah, false, bh,
                                                     (short)0, acc, false, false);
        acc = __builtin_amdgcn_wmma_f32_16x16x32_f16(false, ah, false, bl,
                                                     (short)0, acc, false, false);
        acc = __builtin_amdgcn_wmma_f32_16x16x32_f16(false, al, false, bh,
                                                     (short)0, acc, false, false);
    }

    // C/D layout: element i of v8f -> row M = 8*g + i, column N = rl
    const float bias = b[ng];
#pragma unroll
    for (int i = 0; i < 8; ++i) {
        int row = mtile * 16 + 8 * g + i;
        float v = acc[i] + bias;
        out[(size_t)row * D + ng] = v > 0.f ? v : 0.f;
    }
}

// ------------------------------------------------------------------------------
extern "C" void kernel_launch(void* const* d_in, const int* in_sizes, int n_in,
                              void* d_out, int out_size, void* d_ws, size_t ws_size,
                              hipStream_t stream) {
    const float* h   = (const float*)d_in[0];
    const float* d   = (const float*)d_in[1];
    const int*   src = (const int*)d_in[2];
    const int*   dst = (const int*)d_in[3];
    const float* W   = (const float*)d_in[4];
    const float* b   = (const float*)d_in[5];
    float* out = (float*)d_out;

    float* agg = (float*)d_ws;                                    // 25.6 MB
    int*   deg = (int*)((char*)d_ws + (size_t)N_NODES * D * 4);   // 200 KB

    const int n4 = N_NODES * D / 4;
    zero_ws_kernel<<<(n4 + 255) / 256, 256, 0, stream>>>((float4*)agg, deg,
                                                         n4, N_NODES);
    edge_kernel<<<(N_EDGES + 7) / 8, 256, 0, stream>>>(h, d, src, dst, agg, deg);
    gemm_kernel<<<N_NODES / 16, 256, 0, stream>>>(h, agg, deg, W, b, out);
}
